// GPT_25168508354974
// MI455X (gfx1250) — matrix operand order repaired and checked by reading, not compiled, not run on hardware
//
#include <hip/hip_runtime.h>

#define DEV __device__ __forceinline__

typedef __bf16 bf16;
typedef __attribute__((ext_vector_type(16))) __bf16 v16bf;
typedef __attribute__((ext_vector_type(8))) float v8f;
typedef __attribute__((ext_vector_type(4))) unsigned int u32x4;

// ---- model dims (fixed by the reference) ----
constexpr int Vv = 32000, Dd = 768, Hh = 12, Ll = 6;
constexpr int HD = 64, FFd = 3072, Bb = 4, Tt = 1024;
constexpr int Mm = Bb * Tt;  // 4096 activation rows

// =======================================================================
// WMMA fragment helpers  (v_wmma_f32_16x16x32_bf16; layouts per ISA 7.12.2)
// =======================================================================
union FragBF {
  v16bf v;
  u32x4 u[2];
};

// A matrix 16x32 bf16, row-major source, lda in elements.
DEV FragBF load_a_frag(const bf16* base, int lda) {
  int lane = threadIdx.x & 31;
  const bf16* p = base + (size_t)(lane & 15) * lda + (lane >> 4) * 8;
  FragBF f;
  f.u[0] = *(const u32x4*)(p);
  f.u[1] = *(const u32x4*)(p + 16);
  return f;
}

// B matrix 32x16 bf16 from a pre-transposed [N,K] buffer, ldb in elements.
DEV FragBF load_b_frag(const bf16* base, int ldb) {
  int lane = threadIdx.x & 31;
  const bf16* p = base + (size_t)(lane & 15) * ldb + (lane >> 4) * 8;
  FragBF f;
  f.u[0] = *(const u32x4*)(p);
  f.u[1] = *(const u32x4*)(p + 16);
  return f;
}

DEV v8f wmma_bf16(const FragBF& a, const FragBF& b, v8f c) {
  return __builtin_amdgcn_wmma_f32_16x16x32_bf16(false, a.v, false, b.v,
                                                 (short)0, c, false, false);
}

// =======================================================================
// gfx1250 async global->LDS copy (ASYNCcnt-tracked) + waits
// Generic->LDS addrspace conversion is a truncate on amdgcn (LDS address is
// the low 32 bits).
// =======================================================================
DEV void async_copy_b128(void* lds_dst, const void* gsrc) {
  unsigned l = (unsigned)(unsigned long long)lds_dst;
  asm volatile("global_load_async_to_lds_b128 %0, %1, off" ::"v"(l), "v"(gsrc)
               : "memory");
}
DEV void wait_asynccnt_0() { asm volatile("s_wait_asynccnt 0x0" ::: "memory"); }
DEV void wait_asynccnt_2() { asm volatile("s_wait_asynccnt 0x2" ::: "memory"); }

// =======================================================================
// Generic bf16 GEMM:  C[M,N](f32) = A[M,K](bf16) @ Bt[N,K]^T (+bias)(+res)
// 256 threads = 8 waves; block tile 256x64; wave tile 32x64.
// B tile (64 rows x 64 K) is staged once per block into LDS via async
// copies, double-buffered (16 WMMA between barriers). Rows padded to 144B
// (36 banks, gcd(36,64)=4 -> 16 distinct 4-bank starts): conflict-free.
// =======================================================================
constexpr int BSTR = 72;  // LDS row stride in elements (144 B)

template <bool BIAS, bool GELU, bool RES, bool OUTF, bool OUTB>
__global__ __launch_bounds__(256) void gemm_kernel(
    const bf16* __restrict__ A, const bf16* __restrict__ Bt,
    const float* __restrict__ bias, const float* __restrict__ res,
    float* __restrict__ Cf, bf16* __restrict__ Cb, int N, int K, int tilesN) {
  __shared__ __align__(16) bf16 lds_b[2][64 * BSTR];

  int bn = blockIdx.x % tilesN;
  int bm = blockIdx.x / tilesN;
  int w = threadIdx.x >> 5;
  int lane = threadIdx.x & 31;
  int m0 = bm * 256 + w * 32;
  int n0 = bn * 64;

  // async staging: 64 rows x 128B -> 512 x 16B chunks, 2 per thread
  int c0 = threadIdx.x;        // chunk ids c0 and c0+256
  int r0 = c0 >> 3, s0c = c0 & 7;
  int r1 = (c0 + 256) >> 3, s1c = (c0 + 256) & 7;
  const bf16* bsrc0 = Bt + (size_t)(n0 + r0) * K + s0c * 8;
  const bf16* bsrc1 = Bt + (size_t)(n0 + r1) * K + s1c * 8;
  bf16* bdst0 = &lds_b[0][0] + r0 * BSTR + s0c * 8;
  bf16* bdst1 = &lds_b[0][0] + r1 * BSTR + s1c * 8;

  v8f acc[2][4];
#pragma unroll
  for (int i = 0; i < 2; ++i)
#pragma unroll
    for (int j = 0; j < 4; ++j) acc[i][j] = {};

  // prologue: stage K-step 0 into buffer 0
  async_copy_b128(bdst0, bsrc0);
  async_copy_b128(bdst1, bsrc1);

  int nsteps = K / 64;
  for (int s = 0; s < nsteps; ++s) {
    int k0 = s * 64;
    int cur = s & 1;
    bool more = (s + 1) < nsteps;
    if (more) {
      int boff = (cur ^ 1) * 64 * BSTR;
      async_copy_b128(bdst0 + boff, bsrc0 + (k0 + 64));
      async_copy_b128(bdst1 + boff, bsrc1 + (k0 + 64));
    }
    // own current-buffer copies complete (async loads finish in order)
    if (more)
      wait_asynccnt_2();
    else
      wait_asynccnt_0();
    __syncthreads();  // all 8 waves' slices of buf[cur] landed

    // prefetch the A stream one K-step ahead
    if (k0 + 64 < K) {
      __builtin_prefetch(A + (size_t)(m0 + (lane & 15)) * K + k0 + 64, 0, 0);
      __builtin_prefetch(A + (size_t)(m0 + 16 + (lane & 15)) * K + k0 + 64, 0,
                         0);
    }

#pragma unroll
    for (int kk = 0; kk < 64; kk += 32) {
      FragBF a0 = load_a_frag(A + (size_t)m0 * K + k0 + kk, K);
      FragBF a1 = load_a_frag(A + ((size_t)m0 + 16) * K + k0 + kk, K);
#pragma unroll
      for (int j = 0; j < 4; ++j) {
        FragBF b = load_b_frag(&lds_b[cur][0] + j * 16 * BSTR + kk, BSTR);
        acc[0][j] = wmma_bf16(a0, b, acc[0][j]);
        acc[1][j] = wmma_bf16(a1, b, acc[1][j]);
      }
    }
    __syncthreads();  // readers done before buf[cur] is overwritten
  }

  int col = lane & 15, hi = lane >> 4;
#pragma unroll
  for (int i = 0; i < 2; ++i) {
#pragma unroll
    for (int j = 0; j < 4; ++j) {
      int n = n0 + j * 16 + col;
      float bv = BIAS ? bias[n] : 0.0f;
#pragma unroll
      for (int r = 0; r < 8; ++r) {
        int m = m0 + i * 16 + r + hi * 8;
        float vv = acc[i][j][r] + bv;
        if (GELU) vv = 0.5f * vv * (1.0f + erff(vv * 0.70710678118f));
        if (RES) vv += res[(size_t)m * N + n];
        if (OUTF) Cf[(size_t)m * N + n] = vv;
        if (OUTB) Cb[(size_t)m * N + n] = (bf16)vv;
      }
    }
  }
}

// =======================================================================
// Flash-style causal attention. One wave (32 thr) = one (b,h, 16-query)
// block; online softmax over key tiles of 32; S and P*V both on WMMA.
// q,k: [B*H, T, HD] bf16 ; vT: [B*H, HD, T] bf16 ; out o: [B*T, D] bf16
// =======================================================================
__global__ __launch_bounds__(32) void attention_kernel(
    const bf16* __restrict__ qb, const bf16* __restrict__ kb,
    const bf16* __restrict__ vtb, bf16* __restrict__ ob) {
  int qblk = blockIdx.x & 63;  // T/16 = 64
  int bh = blockIdx.x >> 6;
  int q0 = qblk * 16;
  int lane = threadIdx.x & 31;
  int colL = lane & 15, kh = lane >> 4;

  const bf16* qbase = qb + (size_t)bh * Tt * HD;
  const bf16* kbase = kb + (size_t)bh * Tt * HD;
  const bf16* vbase = vtb + (size_t)bh * HD * Tt;

  FragBF aq0 = load_a_frag(qbase + (size_t)q0 * HD, HD);       // K 0..31
  FragBF aq1 = load_a_frag(qbase + (size_t)q0 * HD + 32, HD);  // K 32..63

  v8f o[4];
#pragma unroll
  for (int t = 0; t < 4; ++t) o[t] = {};
  float mrow[8], lrow[8];
#pragma unroll
  for (int r = 0; r < 8; ++r) {
    mrow[r] = -__builtin_inff();
    lrow[r] = 0.0f;
  }

  __shared__ __align__(16) bf16 lds_p[16 * 32];  // P tile staging (1 KB)

  const float scale = 0.125f;  // 1/sqrt(64)
  int nk32 = (q0 + 16 + 31) / 32;
  for (int kt = 0; kt < nk32; ++kt) {
    int key0 = kt * 32;
    // ---- S = Q @ K^T for 16q x 32k (two 16x16 tiles) ----
    v8f s[2];
    s[0] = {};
    s[1] = {};
#pragma unroll
    for (int j = 0; j < 2; ++j) {
      FragBF b0 = load_b_frag(kbase + (size_t)(key0 + j * 16) * HD, HD);
      FragBF b1 = load_b_frag(kbase + (size_t)(key0 + j * 16) * HD + 32, HD);
      s[j] = wmma_bf16(aq0, b0, s[j]);
      s[j] = wmma_bf16(aq1, b1, s[j]);
    }
    // ---- mask + online softmax (row stats across 16-lane groups) ----
    float alpha[8];
#pragma unroll
    for (int r = 0; r < 8; ++r) {
      int qrow = q0 + r + 8 * kh;
      float s0 = s[0][r] * scale;
      float s1 = s[1][r] * scale;
      if (key0 + colL > qrow) s0 = -__builtin_inff();
      if (key0 + 16 + colL > qrow) s1 = -__builtin_inff();
      float tm = fmaxf(s0, s1);
      tm = fmaxf(tm, __shfl_xor(tm, 1));
      tm = fmaxf(tm, __shfl_xor(tm, 2));
      tm = fmaxf(tm, __shfl_xor(tm, 4));
      tm = fmaxf(tm, __shfl_xor(tm, 8));
      float mn = fmaxf(mrow[r], tm);
      float p0 = __expf(s0 - mn);
      float p1 = __expf(s1 - mn);
      float ts = p0 + p1;
      ts += __shfl_xor(ts, 1);
      ts += __shfl_xor(ts, 2);
      ts += __shfl_xor(ts, 4);
      ts += __shfl_xor(ts, 8);
      alpha[r] = __expf(mrow[r] - mn);
      lrow[r] = lrow[r] * alpha[r] + ts;
      mrow[r] = mn;
      int prow = r + 8 * kh;
      lds_p[prow * 32 + colL] = (bf16)p0;
      lds_p[prow * 32 + 16 + colL] = (bf16)p1;
    }
    // rescale O accumulators by alpha (per-row)
#pragma unroll
    for (int t = 0; t < 4; ++t)
#pragma unroll
      for (int r = 0; r < 8; ++r) o[t][r] *= alpha[r];

    asm volatile("s_wait_dscnt 0" ::: "memory");

    // re-load P in A-fragment layout from LDS (16x32, lda=32)
    const bf16* pp = &lds_p[(lane & 15) * 32 + kh * 8];
    FragBF ap;
    ap.u[0] = *(const u32x4*)(pp);
    ap.u[1] = *(const u32x4*)(pp + 16);

    // ---- O += P @ V  (V^T is [HD,T], contiguous over keys) ----
#pragma unroll
    for (int t = 0; t < 4; ++t) {
      FragBF bv = load_b_frag(vbase + (size_t)(t * 16) * Tt + key0, Tt);
      o[t] = wmma_bf16(ap, bv, o[t]);
    }
  }

  // ---- normalize and write to o[B*T, D] bf16 ----
  int b = bh / Hh, h = bh % Hh;
#pragma unroll
  for (int r = 0; r < 8; ++r) {
    float inv = 1.0f / lrow[r];
    int tok = b * Tt + q0 + r + 8 * kh;
#pragma unroll
    for (int t = 0; t < 4; ++t)
      ob[(size_t)tok * Dd + h * HD + t * 16 + colL] = (bf16)(o[t][r] * inv);
  }
}

// =======================================================================
// LayerNorm: one 256-thread block per row of 768; writes bf16 for WMMA A
// =======================================================================
__global__ __launch_bounds__(256) void layernorm_kernel(
    const float* __restrict__ x, const float* __restrict__ sc,
    const float* __restrict__ bi, bf16* __restrict__ out) {
  int row = blockIdx.x;
  int tid = threadIdx.x;
  const float* xr = x + (size_t)row * Dd;
  float s = 0.0f, s2 = 0.0f;
  for (int i = tid; i < Dd; i += 256) {
    float v = xr[i];
    s += v;
    s2 += v * v;
  }
  __shared__ float red0[256], red1[256];
  red0[tid] = s;
  red1[tid] = s2;
  __syncthreads();
  for (int off = 128; off > 0; off >>= 1) {
    if (tid < off) {
      red0[tid] += red0[tid + off];
      red1[tid] += red1[tid + off];
    }
    __syncthreads();
  }
  float mean = red0[0] * (1.0f / Dd);
  float var = red1[0] * (1.0f / Dd) - mean * mean;
  float rstd = rsqrtf(var + 1e-5f);
  for (int i = tid; i < Dd; i += 256)
    out[(size_t)row * Dd + i] = (bf16)((xr[i] - mean) * rstd * sc[i] + bi[i]);
}

// =======================================================================
// Small elementwise kernels
// =======================================================================
__global__ __launch_bounds__(256) void embed_kernel(
    const int* __restrict__ ids, const float* __restrict__ tok,
    const float* __restrict__ pos, float* __restrict__ x) {
  int idx = blockIdx.x * 256 + threadIdx.x;
  if (idx >= Mm * Dd) return;
  int d = idx % Dd;
  int bt = idx / Dd;
  int t = bt & (Tt - 1);
  x[idx] = tok[(size_t)ids[bt] * Dd + d] + pos[(size_t)t * Dd + d];
}

// qkv f32 [B*T, 3D] (bias already added) -> q,k [BH,T,HD] bf16, vT [BH,HD,T]
__global__ __launch_bounds__(256) void split_qkv_kernel(
    const float* __restrict__ qkv, bf16* __restrict__ qb,
    bf16* __restrict__ kb, bf16* __restrict__ vt) {
  int idx = blockIdx.x * 256 + threadIdx.x;
  if (idx >= Mm * Dd) return;
  int c = idx % Dd;
  int bt = idx / Dd;
  int b = bt >> 10, t = bt & (Tt - 1);
  int h = c >> 6, hd = c & (HD - 1);
  const float* row = qkv + (size_t)bt * (3 * Dd);
  size_t bh = (size_t)(b * Hh + h);
  qb[(bh * Tt + t) * HD + hd] = (bf16)row[c];
  kb[(bh * Tt + t) * HD + hd] = (bf16)row[Dd + c];
  vt[(bh * HD + hd) * Tt + t] = (bf16)row[2 * Dd + c];
}

// W f32 [K,N] -> Wt bf16 [N,K]   (coalesced writes)
__global__ __launch_bounds__(256) void wtrans_kernel(
    const float* __restrict__ W, bf16* __restrict__ Wt, int K, int N) {
  size_t idx = (size_t)blockIdx.x * 256 + threadIdx.x;
  if (idx >= (size_t)K * N) return;
  int k = (int)(idx % K);
  int n = (int)(idx / K);
  Wt[idx] = (bf16)W[(size_t)k * N + n];
}

__global__ __launch_bounds__(256) void cvt_bf16_kernel(
    const float* __restrict__ in, bf16* __restrict__ out, size_t n) {
  size_t idx = (size_t)blockIdx.x * 256 + threadIdx.x;
  if (idx < n) out[idx] = (bf16)in[idx];
}

// =======================================================================
// Host orchestration
// =======================================================================
extern "C" void kernel_launch(void* const* d_in, const int* in_sizes, int n_in,
                              void* d_out, int out_size, void* d_ws,
                              size_t ws_size, hipStream_t stream) {
  const int* ids = (const int*)d_in[0];
  const float* tok = (const float*)d_in[1];
  const float* pos = (const float*)d_in[2];
  const float* ln1s = (const float*)d_in[3];
  const float* ln1b = (const float*)d_in[4];
  const float* qkvw = (const float*)d_in[5];
  const float* qkvb = (const float*)d_in[6];
  const float* outw = (const float*)d_in[7];
  const float* outb = (const float*)d_in[8];
  const float* ln2s = (const float*)d_in[9];
  const float* ln2b = (const float*)d_in[10];
  const float* ff1w = (const float*)d_in[11];
  const float* ff1b = (const float*)d_in[12];
  const float* ff2w = (const float*)d_in[13];
  const float* ff2b = (const float*)d_in[14];
  const float* lnfs = (const float*)d_in[15];
  const float* lnfb = (const float*)d_in[16];
  float* logits = (float*)d_out;

  char* p = (char*)d_ws;
  auto alloc = [&](size_t bytes) -> void* {
    void* r = (void*)p;
    p += (bytes + 255) & ~(size_t)255;
    return r;
  };
  bf16* wt_qkv = (bf16*)alloc((size_t)Ll * 3 * Dd * Dd * 2);
  bf16* wt_out = (bf16*)alloc((size_t)Ll * Dd * Dd * 2);
  bf16* wt_ff1 = (bf16*)alloc((size_t)Ll * FFd * Dd * 2);
  bf16* wt_ff2 = (bf16*)alloc((size_t)Ll * Dd * FFd * 2);
  bf16* tok_bf = (bf16*)alloc((size_t)Vv * Dd * 2);
  float* x = (float*)alloc((size_t)Mm * Dd * 4);
  bf16* h_bf = (bf16*)alloc((size_t)Mm * Dd * 2);
  float* qkv_f = (float*)alloc((size_t)Mm * 3 * Dd * 4);
  bf16* q_bf = (bf16*)alloc((size_t)Mm * Dd * 2);
  bf16* k_bf = (bf16*)alloc((size_t)Mm * Dd * 2);
  bf16* vt_bf = (bf16*)alloc((size_t)Mm * Dd * 2);
  bf16* o_bf = (bf16*)alloc((size_t)Mm * Dd * 2);
  bf16* h1_bf = (bf16*)alloc((size_t)Mm * FFd * 2);
  bf16* xf_bf = (bf16*)alloc((size_t)Mm * Dd * 2);

  auto gblk = [](size_t n) { return (int)((n + 255) / 256); };

  // ---- prep: weights -> transposed bf16 ; tok_emb -> bf16 ----
  for (int l = 0; l < Ll; ++l) {
    wtrans_kernel<<<gblk((size_t)Dd * 3 * Dd), 256, 0, stream>>>(
        qkvw + (size_t)l * Dd * 3 * Dd, wt_qkv + (size_t)l * 3 * Dd * Dd, Dd,
        3 * Dd);
    wtrans_kernel<<<gblk((size_t)Dd * Dd), 256, 0, stream>>>(
        outw + (size_t)l * Dd * Dd, wt_out + (size_t)l * Dd * Dd, Dd, Dd);
    wtrans_kernel<<<gblk((size_t)Dd * FFd), 256, 0, stream>>>(
        ff1w + (size_t)l * Dd * FFd, wt_ff1 + (size_t)l * FFd * Dd, Dd, FFd);
    wtrans_kernel<<<gblk((size_t)FFd * Dd), 256, 0, stream>>>(
        ff2w + (size_t)l * FFd * Dd, wt_ff2 + (size_t)l * Dd * FFd, FFd, Dd);
  }
  cvt_bf16_kernel<<<gblk((size_t)Vv * Dd), 256, 0, stream>>>(tok, tok_bf,
                                                             (size_t)Vv * Dd);

  // ---- embed ----
  embed_kernel<<<gblk((size_t)Mm * Dd), 256, 0, stream>>>(ids, tok, pos, x);

  const int MB = Mm / 256;  // 16 M-tiles of 256 rows
  for (int l = 0; l < Ll; ++l) {
    // LN1 -> h
    layernorm_kernel<<<Mm, 256, 0, stream>>>(x, ln1s + (size_t)l * Dd,
                                             ln1b + (size_t)l * Dd, h_bf);
    // QKV: [4096,768] @ [768,2304] + b
    {
      int tiles = (3 * Dd) / 64;
      gemm_kernel<true, false, false, true, false>
          <<<MB * tiles, 256, 0, stream>>>(
              h_bf, wt_qkv + (size_t)l * 3 * Dd * Dd, qkvb + (size_t)l * 3 * Dd,
              nullptr, qkv_f, nullptr, 3 * Dd, Dd, tiles);
    }
    split_qkv_kernel<<<gblk((size_t)Mm * Dd), 256, 0, stream>>>(qkv_f, q_bf,
                                                                k_bf, vt_bf);
    // causal attention, flash-style
    attention_kernel<<<Bb * Hh * (Tt / 16), 32, 0, stream>>>(q_bf, k_bf, vt_bf,
                                                             o_bf);
    // out proj + residual: x = x + o @ Wo + b
    {
      int tiles = Dd / 64;
      gemm_kernel<true, false, true, true, false>
          <<<MB * tiles, 256, 0, stream>>>(
              o_bf, wt_out + (size_t)l * Dd * Dd, outb + (size_t)l * Dd, x, x,
              nullptr, Dd, Dd, tiles);
    }
    // LN2 -> h
    layernorm_kernel<<<Mm, 256, 0, stream>>>(x, ln2s + (size_t)l * Dd,
                                             ln2b + (size_t)l * Dd, h_bf);
    // FF1 + GELU -> h1 (bf16)
    {
      int tiles = FFd / 64;
      gemm_kernel<true, true, false, false, true>
          <<<MB * tiles, 256, 0, stream>>>(
              h_bf, wt_ff1 + (size_t)l * FFd * Dd, ff1b + (size_t)l * FFd,
              nullptr, nullptr, h1_bf, FFd, Dd, tiles);
    }
    // FF2 + residual: x = x + h1 @ W2 + b
    {
      int tiles = Dd / 64;
      gemm_kernel<true, false, true, true, false>
          <<<MB * tiles, 256, 0, stream>>>(
              h1_bf, wt_ff2 + (size_t)l * Dd * FFd, ff2b + (size_t)l * Dd, x, x,
              nullptr, Dd, FFd, tiles);
    }
  }

  // final LN -> xf ; logits = xf @ tok_emb^T  (tok_bf is already [V, D])
  layernorm_kernel<<<Mm, 256, 0, stream>>>(x, lnfs, lnfb, xf_bf);
  {
    int tiles = Vv / 64;
    gemm_kernel<false, false, false, true, false>
        <<<MB * tiles, 256, 0, stream>>>(xf_bf, tok_bf, nullptr, nullptr,
                                         logits, nullptr, Vv, Dd, tiles);
  }
}
